// WeBGNNDecoder_21388937134519
// MI455X (gfx1250) — matrix operand order repaired and checked by reading, not compiled, not run on hardware
//
#include <hip/hip_runtime.h>
#include <hip/hip_bf16.h>

#define NODE_NUM 2500
#define BATCH    16
#define EDGES_PER 50000
#define GENE_E   48000
#define N_NODES  (BATCH * NODE_NUM)   // 40000
#define E_TOT    (BATCH * EDGES_PER)  // 800000
#define IN_DIM   8
#define HID      128
#define FEAT     384                  // 3 * HID (cat width, also GEMM out width)

typedef _Float16 v16h __attribute__((ext_vector_type(16)));
typedef float    v8f  __attribute__((ext_vector_type(8)));

// ---------------------------------------------------------------- utilities
__global__ void zero_f32_kernel(float* __restrict__ p, int n) {
    for (int i = blockIdx.x * blockDim.x + threadIdx.x; i < n; i += gridDim.x * blockDim.x)
        p[i] = 0.0f;
}

// deg_up[c] = #edges with col==c ; deg_dn[r] = #edges with row==r
__global__ void degree_kernel(const int* __restrict__ row, const int* __restrict__ col,
                              float* __restrict__ du, float* __restrict__ dd) {
    int e = blockIdx.x * blockDim.x + threadIdx.x;
    if (e >= E_TOT) return;
    atomicAdd(&du[col[e]], 1.0f);
    atomicAdd(&dd[row[e]], 1.0f);
}

// x (N x 8, f32) -> hf16 (N x 384, f16), cols >= 8 zero-padded
__global__ void pack_x_kernel(const float* __restrict__ x, _Float16* __restrict__ hf) {
    int idx = blockIdx.x * blockDim.x + threadIdx.x;
    if (idx >= N_NODES * FEAT) return;
    int c = idx % FEAT;
    int i = idx / FEAT;
    hf[idx] = (c < IN_DIM) ? (_Float16)x[i * IN_DIM + c] : (_Float16)0.0f;
}

// Pack [Wu;Wd;Wb] (each HID x K, f32) into one (FEAT x FEAT) f16 block, zero-padded in K.
__global__ void pack_w_kernel(const float* __restrict__ Wu, const float* __restrict__ Wd,
                              const float* __restrict__ Wb, _Float16* __restrict__ Wf, int K) {
    int idx = blockIdx.x * blockDim.x + threadIdx.x;
    if (idx >= FEAT * FEAT) return;
    int r = idx / FEAT, c = idx % FEAT;
    float v = 0.0f;
    if (c < K) {
        const float* src = (r < HID) ? Wu : (r < 2 * HID) ? Wd : Wb;
        v = src[(r % HID) * K + c];
    }
    Wf[idx] = (_Float16)v;
}

// ------------------------------------------------------ WMMA GEMM (wave32)
// out(N x FEAT, f32) = A(N x FEAT, f16) @ Wf(FEAT x FEAT, f16)^T over kSteps*32 K.
// One wave computes one 16x16 tile of out. Fragment layouts per CDNA5 ISA 7.12.2.
__global__ void gemm_wmma_kernel(const _Float16* __restrict__ A,
                                 const _Float16* __restrict__ Wf,
                                 float* __restrict__ out, int kSteps) {
    const int tilesN = FEAT / 16;                       // 24
    int wave = (blockIdx.x * blockDim.x + threadIdx.x) >> 5;
    int lane = threadIdx.x & 31;
    int tm = wave / tilesN;
    int tn = wave % tilesN;
    if (tm >= N_NODES / 16) return;

    int l15 = lane & 15;
    int hi  = lane >> 4;                                // 0 or 1
    const _Float16* arow = A  + (size_t)(tm * 16 + l15) * FEAT;
    const _Float16* brow = Wf + (size_t)(tn * 16 + l15) * FEAT;

    v8f acc = {};
    for (int ks = 0; ks < kSteps; ++ks) {
        int k0 = ks * 32;
        v16h af, bf;
        // A lane layout: K = {k0+hi*8 .. +7} and {k0+16+hi*8 .. +7}
        const _Float16* ap = arow + k0 + hi * 8;
        ((uint4*)&af)[0] = *(const uint4*)(ap);
        ((uint4*)&af)[1] = *(const uint4*)(ap + 16);
        // B lane layout: K = {k0+hi*8 .. +7} and {k0+16+hi*8 .. +7}
        const _Float16* bp = brow + k0 + hi * 8;
        ((uint4*)&bf)[0] = *(const uint4*)(bp);
        ((uint4*)&bf)[1] = *(const uint4*)(bp + 16);
        acc = __builtin_amdgcn_wmma_f32_16x16x32_f16(false, af, false, bf,
                                                     (short)0, acc, false, false);
    }
    // C/D layout: VGPR r, lanes<16 -> M=r, lanes>=16 -> M=8+r; N = lane&15
    int n = tn * 16 + l15;
    int m0 = tm * 16 + hi * 8;
    #pragma unroll
    for (int r = 0; r < 8; ++r)
        out[(size_t)(m0 + r) * FEAT + n] = acc[r];
}

// cat: [0:128)=0 (x_up accum), [128:256)=0 (x_dn accum), [256:384)=bias_x copy
__global__ void cat_init_kernel(const float* __restrict__ lin, float* __restrict__ cat) {
    int idx = blockIdx.x * blockDim.x + threadIdx.x;
    if (idx >= N_NODES * FEAT) return;
    int f = idx % FEAT;
    cat[idx] = (f < 2 * HID) ? 0.0f : lin[idx];
}

// one wave per edge; lane handles 4 features of the 128
__global__ void edge_agg_kernel(const int* __restrict__ row, const int* __restrict__ col,
                                const float* __restrict__ geu, const float* __restrict__ ged,
                                const float* __restrict__ du, const float* __restrict__ dd,
                                const float* __restrict__ lin, float* __restrict__ cat) {
    int t = blockIdx.x * blockDim.x + threadIdx.x;
    int e = t >> 5;
    if (e >= E_TOT) return;
    int lane = t & 31;
    int r = row[e], c = col[e];
    int j = e % EDGES_PER;                               // position within one graph copy
    float eu = (j < GENE_E) ? geu[j] : 1.0f;
    float ed = (j < GENE_E) ? ged[j] : 1.0f;
    float wu = eu / du[c];
    float wd = ed / dd[r];
    int f0 = lane * 4;
    const float4 up = *(const float4*)(lin + (size_t)r * FEAT + f0);          // up_x[row]
    const float4 dn = *(const float4*)(lin + (size_t)c * FEAT + HID + f0);    // dn_x[col]
    float* cu = cat + (size_t)c * FEAT + f0;
    float* cd = cat + (size_t)r * FEAT + HID + f0;
    atomicAdd(cu + 0, wu * up.x); atomicAdd(cu + 1, wu * up.y);
    atomicAdd(cu + 2, wu * up.z); atomicAdd(cu + 3, wu * up.w);
    atomicAdd(cd + 0, wd * dn.x); atomicAdd(cd + 1, wd * dn.y);
    atomicAdd(cd + 2, wd * dn.z); atomicAdd(cd + 3, wd * dn.w);
}

// one wave per node: L2-normalize 384 feats, leaky-relu(0.1), write f32 (in place) + f16 copy
__global__ void norm_lrelu_kernel(float* __restrict__ cat, _Float16* __restrict__ hf) {
    int wave = (blockIdx.x * blockDim.x + threadIdx.x) >> 5;
    if (wave >= N_NODES) return;
    int lane = threadIdx.x & 31;
    float* rp = cat + (size_t)wave * FEAT;
    float v[FEAT / 32];
    float ss = 0.0f;
    #pragma unroll
    for (int i = 0; i < FEAT / 32; ++i) {
        v[i] = rp[lane + i * 32];
        ss += v[i] * v[i];
    }
    #pragma unroll
    for (int m = 16; m >= 1; m >>= 1) ss += __shfl_xor(ss, m, 32);
    float scale = 1.0f / fmaxf(sqrtf(ss), 1e-12f);
    #pragma unroll
    for (int i = 0; i < FEAT / 32; ++i) {
        float y = v[i] * scale;
        y = (y > 0.0f) ? y : 0.1f * y;
        rp[lane + i * 32] = y;
        hf[(size_t)wave * FEAT + lane + i * 32] = (_Float16)y;
    }
}

// decoder: u=a@P1, w=b@P1, v=u@P2, ypred[i]=dot(v[i],w[i]); single block, 512 threads
__global__ void decoder_kernel(const float* __restrict__ h, const int* __restrict__ didx,
                               const float* __restrict__ P1, const float* __restrict__ P2,
                               float* __restrict__ out) {
    __shared__ float u[BATCH][HID];
    __shared__ float w[BATCH][HID];
    __shared__ float vv[BATCH][HID];
    int tid = threadIdx.x;
    for (int idx = tid; idx < BATCH * HID; idx += blockDim.x) {
        int i = idx / HID, p = idx % HID;
        size_t ra = (size_t)(didx[2 * i + 0] - 1) * FEAT;
        size_t rb = (size_t)(didx[2 * i + 1] - 1) * FEAT;
        float su = 0.0f, sw = 0.0f;
        for (int k = 0; k < FEAT; ++k) {
            float p1 = P1[k * HID + p];
            su += h[ra + k] * p1;
            sw += h[rb + k] * p1;
        }
        u[i][p] = su; w[i][p] = sw;
    }
    __syncthreads();
    for (int idx = tid; idx < BATCH * HID; idx += blockDim.x) {
        int i = idx / HID, p = idx % HID;
        float s = 0.0f;
        for (int k = 0; k < HID; ++k) s += u[i][k] * P2[k * HID + p];
        vv[i][p] = s;
    }
    __syncthreads();
    if (tid < BATCH * 32) {
        int i = tid >> 5, lane = tid & 31;
        float s = 0.0f;
        for (int k = lane; k < HID; k += 32) s += vv[i][k] * w[i][k];
        #pragma unroll
        for (int m = 16; m >= 1; m >>= 1) s += __shfl_xor(s, m, 32);
        if (lane == 0) out[i] = s;
    }
}

// ---------------------------------------------------------------- launcher
extern "C" void kernel_launch(void* const* d_in, const int* in_sizes, int n_in,
                              void* d_out, int out_size, void* d_ws, size_t ws_size,
                              hipStream_t stream) {
    const float* x    = (const float*)d_in[0];
    const int*   ei   = (const int*)d_in[1];
    const int*   row  = ei;
    const int*   col  = ei + E_TOT;
    const int*   didx = (const int*)d_in[2];
    const float* Wu[3] = {(const float*)d_in[4],  (const float*)d_in[9],  (const float*)d_in[14]};
    const float* Wd[3] = {(const float*)d_in[5],  (const float*)d_in[10], (const float*)d_in[15]};
    const float* Wb[3] = {(const float*)d_in[6],  (const float*)d_in[11], (const float*)d_in[16]};
    const float* geu[3] = {(const float*)d_in[7],  (const float*)d_in[12], (const float*)d_in[17]};
    const float* ged[3] = {(const float*)d_in[8],  (const float*)d_in[13], (const float*)d_in[18]};
    const float* P1 = (const float*)d_in[19];
    const float* P2 = (const float*)d_in[20];

    char* ws = (char*)d_ws;
    size_t off = 0;
    _Float16* hf16 = (_Float16*)(ws + off); off += (size_t)N_NODES * FEAT * sizeof(_Float16); // 30.72 MB
    float*    lin  = (float*)(ws + off);    off += (size_t)N_NODES * FEAT * sizeof(float);    // 61.44 MB
    float*    cat  = (float*)(ws + off);    off += (size_t)N_NODES * FEAT * sizeof(float);    // 61.44 MB
    float*    degs = (float*)(ws + off);    off += (size_t)2 * N_NODES * sizeof(float);       // du | dd
    float*    du = degs;
    float*    dd = degs + N_NODES;
    _Float16* Wf = (_Float16*)(ws + off);   off += (size_t)FEAT * FEAT * sizeof(_Float16);

    const int NT = 256;
    // degrees (edge structure only; recomputed each call for determinism)
    zero_f32_kernel<<<(2 * N_NODES + NT - 1) / NT, NT, 0, stream>>>(degs, 2 * N_NODES);
    degree_kernel<<<(E_TOT + NT - 1) / NT, NT, 0, stream>>>(row, col, du, dd);

    // h0 = x (f16, zero padded to FEAT)
    pack_x_kernel<<<(N_NODES * FEAT + NT - 1) / NT, NT, 0, stream>>>(x, hf16);

    const int gemmTiles  = (N_NODES / 16) * (FEAT / 16);           // 60000 wave-tiles
    const int gemmBlocks = (gemmTiles + (NT / 32) - 1) / (NT / 32);
    const int elemBlocks = (N_NODES * FEAT + NT - 1) / NT;
    const int edgeBlocks = ((E_TOT * 32) + NT - 1) / NT;
    const int nodeBlocks = ((N_NODES * 32) + NT - 1) / NT;

    for (int l = 0; l < 3; ++l) {
        int K = (l == 0) ? IN_DIM : FEAT;
        int kSteps = (l == 0) ? 1 : (FEAT / 32);                   // K padded to 32 for layer 1
        pack_w_kernel<<<(FEAT * FEAT + NT - 1) / NT, NT, 0, stream>>>(Wu[l], Wd[l], Wb[l], Wf, K);
        gemm_wmma_kernel<<<gemmBlocks, NT, 0, stream>>>(hf16, Wf, lin, kSteps);
        cat_init_kernel<<<elemBlocks, NT, 0, stream>>>(lin, cat);
        edge_agg_kernel<<<edgeBlocks, NT, 0, stream>>>(row, col, geu[l], ged[l], du, dd, lin, cat);
        norm_lrelu_kernel<<<nodeBlocks, NT, 0, stream>>>(cat, hf16);
    }

    decoder_kernel<<<1, 512, 0, stream>>>(cat, didx, P1, P2, (float*)d_out);
}
